// SoftModularizationNetwork_33595234189705
// MI455X (gfx1250) — matrix operand order repaired and checked by reading, not compiled, not run on hardware
//
#include <hip/hip_runtime.h>
#include <hip/hip_bf16.h>

typedef __bf16 bf16;
typedef __attribute__((ext_vector_type(16))) __bf16 bf16x16;
typedef __attribute__((ext_vector_type(8)))  __bf16 bf16x8;
typedef __attribute__((ext_vector_type(8)))  float  f32x8;
typedef __attribute__((ext_vector_type(4)))  int    v4i;

union Frag16 { bf16x16 v; bf16x8 h[2]; };

#define ASTRIDE 40   // padded LDS row stride (keeps 16B align for b128, shifts banks)
#define BSTRIDE 40

// ---- gfx1250 async global->LDS (ASYNCcnt) --------------------------------
#if defined(__has_builtin)
#  if __has_builtin(__builtin_amdgcn_global_load_async_to_lds_b128)
#    define USE_ASYNC_LDS 1
#  endif
#  if __has_builtin(__builtin_amdgcn_s_wait_asynccnt)
#    define HAVE_WAIT_ASYNC 1
#  endif
#endif
#ifndef USE_ASYNC_LDS
#  define USE_ASYNC_LDS 0
#endif

#define AS1 __attribute__((address_space(1)))
#define AS3 __attribute__((address_space(3)))

__device__ __forceinline__ void async_wait0()
{
#if USE_ASYNC_LDS
#  if defined(HAVE_WAIT_ASYNC)
    __builtin_amdgcn_s_wait_asynccnt(0);
#  else
    asm volatile("s_wait_asynccnt 0x0" ::: "memory");
#  endif
#endif
}

// ---------------------------------------------------------------------------
// WMMA bf16 GEMM:  Out[M x N] = epilogue( A[M x K] @ Wt[N x K]^T + bias )
//   A row-major bf16 (lda), Wt row-major bf16 (ldw = padded K) i.e. W transposed
//   EPI bit0 = ReLU, bit1 = multiply by Other[row,col] (bf16) after bias
//   OUTF32: 1 -> write float, 0 -> write bf16
//   FULLN:  1 -> every block's 128-col tile fully inside N (no guards, async DMA)
//   Grouped via blockIdx.z with element strides sA/sW/sBias/sOut (sA==0 => broadcast A)
// Block: 256 threads (8 waves), tile 128x128, K-step 32, double-buffered LDS.
// ---------------------------------------------------------------------------
template<int EPI, int OUTF32, int FULLN>
__global__ __launch_bounds__(256) void gemm_bf16_wmma(
    const bf16* __restrict__ A, int lda, long long sA,
    const bf16* __restrict__ Wt, int ldw, long long sW,
    const float* __restrict__ bias, long long sBias,
    void* __restrict__ Out, int ldo, long long sOut,
    const bf16* __restrict__ Other, int ldother,
    int N, int K, int n_real)
{
    __shared__ bf16 As[2][128 * ASTRIDE];
    __shared__ bf16 Bs[2][128 * BSTRIDE];

    const int tid   = threadIdx.x;
    const int lane  = tid & 31;
    const int wid   = tid >> 5;
    const int waveM = wid & 1;    // 2 waves over M (64 rows each)
    const int waveN = wid >> 1;   // 4 waves over N (32 cols each)
    const int z     = blockIdx.z;
    const int mBase = blockIdx.y * 128;
    const int nBase = blockIdx.x * 128;

    A    += z * sA;
    Wt   += z * sW;
    bias += z * sBias;
    float* outF = (float*)Out + z * sOut;
    bf16*  outB = (bf16*)Out  + z * sOut;

    f32x8 acc[4][2];
#pragma unroll
    for (int mt = 0; mt < 4; ++mt)
#pragma unroll
        for (int nt = 0; nt < 2; ++nt)
            acc[mt][nt] = (f32x8)0.0f;

    const int lrow  = lane & 15;
    const int kHalf = lane >> 4;

    // global->LDS mapping: 256 threads, each thread 4x b128 per 128x32 tile pair
    const int gr = tid >> 2;         // 0..63 (rows gr and gr+64)
    const int gc = (tid & 3) * 8;    // 0,8,16,24 (bf16 elems)

    const bf16* gA0 = A  + (long long)(mBase + gr) * lda + gc;
    const bf16* gA1 = gA0 + (long long)64 * lda;
    const bf16* gB0 = Wt + (long long)(nBase + gr) * ldw + gc;
    const bf16* gB1 = gB0 + (long long)64 * ldw;
    const bool bok0 = FULLN || (nBase + gr < N);
    const bool bok1 = FULLN || (nBase + gr + 64 < N);

    auto issue_tile = [&](int buf, int k0) {
#if USE_ASYNC_LDS
        if (FULLN) {
            __builtin_amdgcn_global_load_async_to_lds_b128(
                (AS1 v4i*)(gA0 + k0), (AS3 v4i*)&As[buf][gr * ASTRIDE + gc], 0, 0);
            __builtin_amdgcn_global_load_async_to_lds_b128(
                (AS1 v4i*)(gA1 + k0), (AS3 v4i*)&As[buf][(gr + 64) * ASTRIDE + gc], 0, 0);
            __builtin_amdgcn_global_load_async_to_lds_b128(
                (AS1 v4i*)(gB0 + k0), (AS3 v4i*)&Bs[buf][gr * BSTRIDE + gc], 0, 0);
            __builtin_amdgcn_global_load_async_to_lds_b128(
                (AS1 v4i*)(gB1 + k0), (AS3 v4i*)&Bs[buf][(gr + 64) * BSTRIDE + gc], 0, 0);
            return;
        }
#endif
        bf16x8 a0 = *(const bf16x8*)(gA0 + k0);
        bf16x8 a1 = *(const bf16x8*)(gA1 + k0);
        bf16x8 b0 = {}; if (bok0) b0 = *(const bf16x8*)(gB0 + k0);
        bf16x8 b1 = {}; if (bok1) b1 = *(const bf16x8*)(gB1 + k0);
        *(bf16x8*)&As[buf][gr * ASTRIDE + gc]        = a0;
        *(bf16x8*)&As[buf][(gr + 64) * ASTRIDE + gc] = a1;
        *(bf16x8*)&Bs[buf][gr * BSTRIDE + gc]        = b0;
        *(bf16x8*)&Bs[buf][(gr + 64) * BSTRIDE + gc] = b1;
    };

    const int ntiles = K >> 5;
    issue_tile(0, 0);

    for (int t = 0; t < ntiles; ++t) {
        const int cur = t & 1;
        if (FULLN) async_wait0();       // this wave's DMA for tile t complete
        __syncthreads();                // all waves: tile t visible, prior reads of other buf done
        if (t + 1 < ntiles) issue_tile(1 - cur, (t + 1) << 5);  // overlap with compute below

        Frag16 af[4], bfr[2];
#pragma unroll
        for (int mt = 0; mt < 4; ++mt) {
            int m = waveM * 64 + mt * 16 + lrow;
            af[mt].h[0] = *(const bf16x8*)&As[cur][m * ASTRIDE + kHalf * 8];
            af[mt].h[1] = *(const bf16x8*)&As[cur][m * ASTRIDE + kHalf * 8 + 16];
        }
#pragma unroll
        for (int nt = 0; nt < 2; ++nt) {
            int n = waveN * 32 + nt * 16 + lrow;
            bfr[nt].h[0] = *(const bf16x8*)&Bs[cur][n * BSTRIDE + kHalf * 8];
            bfr[nt].h[1] = *(const bf16x8*)&Bs[cur][n * BSTRIDE + kHalf * 8 + 16];
        }
#pragma unroll
        for (int mt = 0; mt < 4; ++mt)
#pragma unroll
            for (int nt = 0; nt < 2; ++nt)
                acc[mt][nt] = __builtin_amdgcn_wmma_f32_16x16x32_bf16(
                    false, af[mt].v, false, bfr[nt].v,
                    (short)0, acc[mt][nt], false, false);
    }

    // epilogue: C/D layout -> VGPR r: lanes0-15 M=r, lanes16-31 M=r+8; N=lane%16
    const int colLane = lane & 15;
    const int rowHalf = (lane >> 4) * 8;
#pragma unroll
    for (int mt = 0; mt < 4; ++mt) {
#pragma unroll
        for (int nt = 0; nt < 2; ++nt) {
            int col = nBase + waveN * 32 + nt * 16 + colLane;
            if (!FULLN && col >= n_real) continue;
            float bv = bias[col];
#pragma unroll
            for (int r = 0; r < 8; ++r) {
                int row = mBase + waveM * 64 + mt * 16 + rowHalf + r;
                float v = acc[mt][nt][r] + bv;
                if (EPI & 2) v *= (float)Other[(long long)row * ldother + col];
                if (EPI & 1) v = v > 0.0f ? v : 0.0f;
                if (OUTF32) outF[(long long)row * ldo + col] = v;
                else        outB[(long long)row * ldo + col] = (bf16)v;
            }
        }
    }
}

// ---------------------------------------------------------------------------
// Split x into zero-padded bf16 obs (B x 1024, cols<974 valid) and task (B x 64, cols<50)
// ---------------------------------------------------------------------------
__global__ void prep_x_kernel(const float* __restrict__ x, bf16* __restrict__ obs,
                              bf16* __restrict__ task, int Bn)
{
    long long i = (long long)blockIdx.x * blockDim.x + threadIdx.x;
    if (i >= (long long)Bn * 1088) return;
    int b = (int)(i / 1088), j = (int)(i % 1088);
    if (j < 1024) {
        obs[(long long)b * 1024 + j] = (j < 974) ? (bf16)x[(long long)b * 1024 + j] : (bf16)0.0f;
    } else {
        int jt = j - 1024;
        task[(long long)b * 64 + jt] = (jt < 50) ? (bf16)x[(long long)b * 1024 + 974 + jt] : (bf16)0.0f;
    }
}

// W (K x N) f32 row-major -> Wt (Npad x Kpad) bf16 row-major (transposed, zero-padded)
__global__ void wt_conv_kernel(const float* __restrict__ W, bf16* __restrict__ Wt,
                               int K, int N, int Kpad, int Npad,
                               long long sW, long long sWt)
{
    long long i = (long long)blockIdx.x * blockDim.x + threadIdx.x;
    if (i >= (long long)Npad * Kpad) return;
    int z = blockIdx.z;
    int n = (int)(i / Kpad), k = (int)(i % Kpad);
    float v = (k < K && n < N) ? W[z * sW + (long long)k * N + n] : 0.0f;
    Wt[z * sWt + (long long)n * Kpad + k] = (bf16)v;
}

__global__ void relu_bf_kernel(const bf16* __restrict__ in, bf16* __restrict__ out, long long n)
{
    long long i = (long long)blockIdx.x * blockDim.x + threadIdx.x;
    if (i >= n) return;
    float v = (float)in[i];
    out[i] = (bf16)(v > 0.0f ? v : 0.0f);
}

// softmax over groups of 8; writes f32 probs and optionally bf16 copy into prev buffer
__global__ void softmax8_kernel(const float* __restrict__ logits, int ldl,
                                float* __restrict__ probs, int ldp,
                                bf16* __restrict__ prevOut, int ldprev,
                                int Bn, int G)
{
    int i = blockIdx.x * blockDim.x + threadIdx.x;
    if (i >= Bn * G) return;
    int b = i / G, g = i % G;
    const float* lp = logits + (long long)b * ldl + g * 8;
    float mx = lp[0];
#pragma unroll
    for (int j = 1; j < 8; ++j) mx = fmaxf(mx, lp[j]);
    float e[8], s = 0.0f;
#pragma unroll
    for (int j = 0; j < 8; ++j) { e[j] = __expf(lp[j] - mx); s += e[j]; }
    float inv = 1.0f / s;
    float* pp = probs + (long long)b * ldp + g * 8;
#pragma unroll
    for (int j = 0; j < 8; ++j) {
        float p = e[j] * inv;
        pp[j] = p;
        if (prevOut) prevOut[(long long)b * ldprev + g * 8 + j] = (bf16)p;
    }
}

// ins[m,b,d] = relu( sum_n probs[b, m*8+n] * dense[n,b,d] )   (dense/ins layout: (M,B,D))
__global__ void mix_kernel(const float* __restrict__ probs, const bf16* __restrict__ dense,
                           bf16* __restrict__ ins, int Bn, int D)
{
    long long i = (long long)blockIdx.x * blockDim.x + threadIdx.x;
    if (i >= (long long)8 * Bn * D) return;
    int d = (int)(i % D);
    long long t = i / D;
    int b = (int)(t % Bn);
    int m = (int)(t / Bn);
    float s = 0.0f;
#pragma unroll
    for (int n = 0; n < 8; ++n)
        s += probs[(long long)b * 64 + m * 8 + n] * (float)dense[((long long)n * Bn + b) * D + d];
    ins[((long long)m * Bn + b) * D + d] = (bf16)(s > 0.0f ? s : 0.0f);
}

// out_e[b,d] = relu( sum_m probs[b,m] * dense[m,b,d] )
__global__ void wsum_kernel(const float* __restrict__ probs, const bf16* __restrict__ dense,
                            bf16* __restrict__ out_e, int Bn, int D)
{
    long long i = (long long)blockIdx.x * blockDim.x + threadIdx.x;
    if (i >= (long long)Bn * D) return;
    int d = (int)(i % D);
    int b = (int)(i / D);
    float s = 0.0f;
#pragma unroll
    for (int m = 0; m < 8; ++m)
        s += probs[(long long)b * 64 + m] * (float)dense[((long long)m * Bn + b) * D + d];
    out_e[(long long)b * D + d] = (bf16)(s > 0.0f ? s : 0.0f);
}

// ---------------------------------------------------------------------------
extern "C" void kernel_launch(void* const* d_in, const int* in_sizes, int n_in,
                              void* d_out, int out_size, void* d_ws, size_t ws_size,
                              hipStream_t stream)
{
    (void)in_sizes; (void)n_in; (void)out_size; (void)ws_size;
    const int Bn = 8192;

    const float* x   = (const float*)d_in[0];
    const float* Wf1 = (const float*)d_in[1];  const float* bf1v = (const float*)d_in[2];
    const float* Wf2 = (const float*)d_in[3];  const float* bf2v = (const float*)d_in[4];
    const float* Wz  = (const float*)d_in[5];  const float* bzv  = (const float*)d_in[6];
    const float* Wt1 = (const float*)d_in[7];  const float* bt1v = (const float*)d_in[8];
    const float* Wt2 = (const float*)d_in[9];  const float* bt2v = (const float*)d_in[10];
    const float* Wm[4]  = {(const float*)d_in[11], (const float*)d_in[13], (const float*)d_in[15], (const float*)d_in[17]};
    const float* bm[4]  = {(const float*)d_in[12], (const float*)d_in[14], (const float*)d_in[16], (const float*)d_in[18]};
    const float* Wpo[4] = {(const float*)d_in[19], (const float*)d_in[21], (const float*)d_in[25], (const float*)d_in[29]};
    const float* bpo[4] = {(const float*)d_in[20], (const float*)d_in[22], (const float*)d_in[26], (const float*)d_in[30]};
    const float* Wpe[4] = {nullptr, (const float*)d_in[23], (const float*)d_in[27], (const float*)d_in[31]};
    const float* bpe[4] = {nullptr, (const float*)d_in[24], (const float*)d_in[28], (const float*)d_in[32]};
    const float* Wout = (const float*)d_in[33]; const float* boutv = (const float*)d_in[34];
    float* out = (float*)d_out;

    char* wp = (char*)d_ws;
    auto alloc = [&](size_t bytes) -> void* {
        void* r = (void*)wp;
        wp += (bytes + 255) & ~(size_t)255;
        return r;
    };

    // activations
    bf16* OBS   = (bf16*)alloc((size_t)Bn * 1024 * 2);   // padded obs; later reused for t2
    bf16* TASK  = (bf16*)alloc((size_t)Bn * 64 * 2);
    bf16* H1    = (bf16*)alloc((size_t)Bn * 1024 * 2);   // h1; later reused for t1
    bf16* OBS_E = (bf16*)alloc((size_t)Bn * 1024 * 2);   // obs_e, relu'd in place after gating use
    bf16* TE    = (bf16*)alloc((size_t)Bn * 1024 * 2);
    bf16* RTE   = (bf16*)alloc((size_t)Bn * 1024 * 2);   // relu(te_i), reused each layer
    bf16* DENSE = (bf16*)alloc((size_t)8 * Bn * 512 * 2);
    bf16* INS   = (bf16*)alloc((size_t)8 * Bn * 512 * 2);
    bf16* PREV  = (bf16*)alloc((size_t)Bn * 192 * 2);
    bf16* OUTE  = (bf16*)alloc((size_t)Bn * 512 * 2);
    float* LOG  = (float*)alloc((size_t)Bn * 64 * 4);
    float* PRB  = (float*)alloc((size_t)Bn * 64 * 4);
    // bf16 transposed weights (N x Kpad)
    bf16* Wf1t  = (bf16*)alloc((size_t)1024 * 1024 * 2);
    bf16* Wf2t  = (bf16*)alloc((size_t)1024 * 1024 * 2);
    bf16* Wzt   = (bf16*)alloc((size_t)1024 * 64 * 2);
    bf16* Wt1t  = (bf16*)alloc((size_t)1024 * 1024 * 2);
    bf16* Wt2t  = (bf16*)alloc((size_t)1024 * 1024 * 2);
    bf16* Wm0t  = (bf16*)alloc((size_t)8 * 512 * 1024 * 2);
    bf16* Wmt[4]; Wmt[0] = Wm0t;
    for (int i = 1; i < 4; ++i) Wmt[i] = (bf16*)alloc((size_t)8 * 512 * 512 * 2);
    bf16* Wpot[4];
    Wpot[0] = (bf16*)alloc((size_t)64 * 1024 * 2);
    Wpot[1] = (bf16*)alloc((size_t)64 * 1024 * 2);
    Wpot[2] = (bf16*)alloc((size_t)64 * 1024 * 2);
    Wpot[3] = (bf16*)alloc((size_t)16 * 1024 * 2);
    bf16* Wpet[4] = {nullptr, nullptr, nullptr, nullptr};
    Wpet[1] = (bf16*)alloc((size_t)1024 * 64 * 2);
    Wpet[2] = (bf16*)alloc((size_t)1024 * 128 * 2);
    Wpet[3] = (bf16*)alloc((size_t)1024 * 192 * 2);
    bf16* Woutt = (bf16*)alloc((size_t)512 * 512 * 2);

    const int CT = 256;
    auto nblk = [&](long long n) { return (unsigned)((n + CT - 1) / CT); };

    // ---- preprocess inputs & weights ----
    prep_x_kernel<<<nblk((long long)Bn * 1088), CT, 0, stream>>>(x, OBS, TASK, Bn);
    wt_conv_kernel<<<nblk(1024LL * 1024), CT, 0, stream>>>(Wf1, Wf1t, 974, 1024, 1024, 1024, 0, 0);
    wt_conv_kernel<<<nblk(1024LL * 1024), CT, 0, stream>>>(Wf2, Wf2t, 1024, 1024, 1024, 1024, 0, 0);
    wt_conv_kernel<<<nblk(1024LL * 64),   CT, 0, stream>>>(Wz,  Wzt,  50, 1024, 64, 1024, 0, 0);
    wt_conv_kernel<<<nblk(1024LL * 1024), CT, 0, stream>>>(Wt1, Wt1t, 1024, 1024, 1024, 1024, 0, 0);
    wt_conv_kernel<<<nblk(1024LL * 1024), CT, 0, stream>>>(Wt2, Wt2t, 1024, 1024, 1024, 1024, 0, 0);
    wt_conv_kernel<<<dim3(nblk(512LL * 1024), 1, 8), CT, 0, stream>>>(Wm[0], Wmt[0], 1024, 512, 1024, 512, 1024LL * 512, 512LL * 1024);
    for (int i = 1; i < 4; ++i)
        wt_conv_kernel<<<dim3(nblk(512LL * 512), 1, 8), CT, 0, stream>>>(Wm[i], Wmt[i], 512, 512, 512, 512, 512LL * 512, 512LL * 512);
    for (int i = 0; i < 3; ++i)
        wt_conv_kernel<<<nblk(64LL * 1024), CT, 0, stream>>>(Wpo[i], Wpot[i], 1024, 64, 1024, 64, 0, 0);
    wt_conv_kernel<<<nblk(16LL * 1024), CT, 0, stream>>>(Wpo[3], Wpot[3], 1024, 8, 1024, 16, 0, 0);
    wt_conv_kernel<<<nblk(1024LL * 64),  CT, 0, stream>>>(Wpe[1], Wpet[1], 64, 1024, 64, 1024, 0, 0);
    wt_conv_kernel<<<nblk(1024LL * 128), CT, 0, stream>>>(Wpe[2], Wpet[2], 128, 1024, 128, 1024, 0, 0);
    wt_conv_kernel<<<nblk(1024LL * 192), CT, 0, stream>>>(Wpe[3], Wpet[3], 192, 1024, 192, 1024, 0, 0);
    wt_conv_kernel<<<nblk(512LL * 512),  CT, 0, stream>>>(Wout, Woutt, 512, 512, 512, 512, 0, 0);

    const dim3 blk(256);
    const dim3 gN1024(8, Bn / 128, 1);
    const dim3 gN64(1, Bn / 128, 1);
    const dim3 gN512(4, Bn / 128, 1);
    const dim3 gMod(4, Bn / 128, 8);

    // ---- trunk ----
    // h1 = relu(obs @ Wf1 + bf1)
    gemm_bf16_wmma<1, 0, 1><<<gN1024, blk, 0, stream>>>(OBS, 1024, 0, Wf1t, 1024, 0, bf1v, 0,
                                                        H1, 1024, 0, nullptr, 0, 1024, 1024, 1024);
    // obs_e = h1 @ Wf2 + bf2
    gemm_bf16_wmma<0, 0, 1><<<gN1024, blk, 0, stream>>>(H1, 1024, 0, Wf2t, 1024, 0, bf2v, 0,
                                                        OBS_E, 1024, 0, nullptr, 0, 1024, 1024, 1024);
    // t1 = relu((task @ Wz + bz) * obs_e)   (reuse H1)
    gemm_bf16_wmma<3, 0, 1><<<gN1024, blk, 0, stream>>>(TASK, 64, 0, Wzt, 64, 0, bzv, 0,
                                                        H1, 1024, 0, OBS_E, 1024, 1024, 64, 1024);
    // obs_e <- relu(obs_e) in place (only relu'd version needed from here on)
    relu_bf_kernel<<<nblk((long long)Bn * 1024), CT, 0, stream>>>(OBS_E, OBS_E, (long long)Bn * 1024);
    // t2 = relu(t1 @ Wt1 + bt1)   (reuse OBS)
    gemm_bf16_wmma<1, 0, 1><<<gN1024, blk, 0, stream>>>(H1, 1024, 0, Wt1t, 1024, 0, bt1v, 0,
                                                        OBS, 1024, 0, nullptr, 0, 1024, 1024, 1024);
    // te = t2 @ Wt2 + bt2
    gemm_bf16_wmma<0, 0, 1><<<gN1024, blk, 0, stream>>>(OBS, 1024, 0, Wt2t, 1024, 0, bt2v, 0,
                                                        TE, 1024, 0, nullptr, 0, 1024, 1024, 1024);
    // rte = relu(te)  (layer-0 policy input)
    relu_bf_kernel<<<nblk((long long)Bn * 1024), CT, 0, stream>>>(TE, RTE, (long long)Bn * 1024);

    const long long sDense = (long long)Bn * 512;
    const int peK[4] = {0, 64, 128, 192};

    for (int i = 0; i < 4; ++i) {
        // dense_i[m] = ins_i[m] @ Wm_i[m] + bm_i[m]   (i==0: A broadcast = relu(obs_e))
        if (i == 0)
            gemm_bf16_wmma<0, 0, 1><<<gMod, blk, 0, stream>>>(OBS_E, 1024, 0, Wmt[0], 1024, 512LL * 1024,
                                                              bm[0], 512, DENSE, 512, sDense,
                                                              nullptr, 0, 512, 1024, 512);
        else
            gemm_bf16_wmma<0, 0, 1><<<gMod, blk, 0, stream>>>(INS, 512, sDense, Wmt[i], 512, 512LL * 512,
                                                              bm[i], 512, DENSE, 512, sDense,
                                                              nullptr, 0, 512, 512, 512);
        // rte_i = relu(te * (prev @ Wpe_i + bpe_i))  for i>0 (i==0 uses relu(te) already in RTE)
        if (i > 0)
            gemm_bf16_wmma<3, 0, 1><<<gN1024, blk, 0, stream>>>(PREV, 192, 0, Wpet[i], peK[i], 0, bpe[i], 0,
                                                                RTE, 1024, 0, TE, 1024, 1024, peK[i], 1024);
        // logits_i = rte_i @ Wpo_i + bpo_i   (f32; layer 3 has n_real=8, padded N=16)
        gemm_bf16_wmma<0, 1, 0><<<gN64, blk, 0, stream>>>(RTE, 1024, 0, Wpot[i], 1024, 0, bpo[i], 0,
                                                          LOG, 64, 0, nullptr, 0,
                                                          (i == 3) ? 16 : 64, 1024, (i == 3) ? 8 : 64);
        if (i < 3) {
            softmax8_kernel<<<nblk((long long)Bn * 8), CT, 0, stream>>>(LOG, 64, PRB, 64,
                                                                        PREV + i * 64, 192, Bn, 8);
            // ins_{i+1}[m,b,:] = relu(sum_n probs[b,m,n] * dense[n,b,:])
            mix_kernel<<<nblk(8LL * Bn * 512), CT, 0, stream>>>(PRB, DENSE, INS, Bn, 512);
        } else {
            softmax8_kernel<<<nblk((long long)Bn), CT, 0, stream>>>(LOG, 64, PRB, 64,
                                                                    nullptr, 0, Bn, 1);
            // out_e = relu(sum_m dense[m,b,:] * probs[b,m])
            wsum_kernel<<<nblk((long long)Bn * 512), CT, 0, stream>>>(PRB, DENSE, OUTE, Bn, 512);
        }
    }

    // out = out_e @ Wout + bout   (f32 -> d_out)
    gemm_bf16_wmma<0, 1, 1><<<gN512, blk, 0, stream>>>(OUTE, 512, 0, Woutt, 512, 0, boutv, 0,
                                                       out, 512, 0, nullptr, 0, 512, 512, 512);
}
